// _hierarchyAttentionFasterRCNN_75531294867804
// MI455X (gfx1250) — matrix operand order repaired and checked by reading, not compiled, not run on hardware
//
#include <hip/hip_runtime.h>

typedef __attribute__((ext_vector_type(16))) __bf16 v16bf;
typedef __attribute__((ext_vector_type(8)))  __bf16 v8bf;
typedef __attribute__((ext_vector_type(8)))  float  v8f;

#define DEVINL static __device__ __forceinline__

// ---- bf16 helpers: native casts (gfx1250 has hw bf16 cvt) ----
DEVINL __bf16 f2bf(float f) { return (__bf16)f; }
DEVINL float  bf2f(__bf16 b) { return (float)b; }

// ---- WMMA wrapper ----
DEVINL v8f wmma_bf16(v16bf a, v16bf b, v8f c) {
    return __builtin_amdgcn_wmma_f32_16x16x32_bf16(
        /*neg_a=*/false, a, /*neg_b=*/false, b,
        /*c_mod=*/(short)0, c, /*reuse_a=*/false, /*reuse_b=*/false);
}

// A fragment (16x32, lane=row M): lanes0-15 hold K {0..15}, lanes16-31 {16..31}.
DEVINL v16bf a_from_bf(const __bf16* lo, const __bf16* hi8) {
    v8bf l = *(const v8bf*)lo;
    v8bf h = *(const v8bf*)hi8;
    v16bf r;
#pragma unroll
    for (int i = 0; i < 8; ++i) { r[i] = l[i]; r[8 + i] = h[i]; }
    return r;
}
DEVINL v16bf a_from_f32(const float* lo, const float* hi8) {
    v16bf r;
#pragma unroll
    for (int i = 0; i < 8; ++i) { r[i] = f2bf(lo[i]); r[8 + i] = f2bf(hi8[i]); }
    return r;
}

// ================= prep: transpose + convert to bf16 =================
__global__ void k_transpose_cvt(const float* __restrict__ in, __bf16* __restrict__ out,
                                int rows, int cols) {
    int idx = blockIdx.x * blockDim.x + threadIdx.x;
    if (idx >= rows * cols) return;
    int c = idx / rows, r = idx % rows;
    out[idx] = f2bf(in[(size_t)r * cols + c]);
}

__global__ void k_copy_cvt(const float* __restrict__ in, __bf16* __restrict__ out, int n) {
    int i = blockIdx.x * blockDim.x + threadIdx.x;
    if (i < n) out[i] = f2bf(in[i]);
}

// ================= position MLP: logpos[n][g][m] =================
__global__ void __launch_bounds__(256)
k_pos(const float* __restrict__ pe, const __bf16* __restrict__ WposT,
      const float* __restrict__ bpos, __bf16* __restrict__ logpos) {
    int w = threadIdx.x >> 5, lane = threadIdx.x & 31;
    int hi = lane >> 4, col = lane & 15;
    int tile = blockIdx.x * 8 + w;        // 65536 tiles total
    int nm0 = tile * 16;
    int n = nm0 >> 10, m0 = nm0 & 1023;

    const float* arow = pe + (size_t)(nm0 + col) * 64;   // row M = lane&15
    v16bf a0 = a_from_f32(arow + hi * 16,       arow + 8 + hi * 16);
    v16bf a1 = a_from_f32(arow + 32 + hi * 16,  arow + 40 + hi * 16);

    const __bf16* bp = WposT + col * 64;                 // col N = group
    v16bf b0 = *(const v16bf*)(bp + hi * 16);
    v16bf b1 = *(const v16bf*)(bp + 32 + hi * 16);

    v8f c = {};
    c = wmma_bf16(a0, b0, c);
    c = wmma_bf16(a1, b1, c);

    float bias = bpos[col];
#pragma unroll
    for (int i = 0; i < 8; ++i) {
        float v = c[i] + bias;
        v = v > 0.f ? v : 0.f;
        v = logf(fmaxf(v, 1e-6f));
        int m = m0 + i + hi * 8;
        logpos[((size_t)n * 16 + col) * 1024 + m] = f2bf(v);
    }
}

// ================= q/k projection: 1024x1024x1024 GEMM =================
// A tile (16 rows x 1024 K) staged once per block in LDS as bf16.
__global__ void __launch_bounds__(256)
k_proj(const float* __restrict__ roi, const __bf16* __restrict__ WT,
       const float* __restrict__ bias, __bf16* __restrict__ outbf) {
    __shared__ __bf16 sA[16 * 1024];     // 32 KB
    int tid = threadIdx.x;
    int w = tid >> 5, lane = tid & 31;
    int hi = lane >> 4, col = lane & 15;
    int n0 = blockIdx.x * 16;
    int c0 = blockIdx.y * 128 + w * 16;

    const float* ablk = roi + (size_t)n0 * 1024;   // rows are contiguous
    for (int idx = tid; idx < 16 * 1024; idx += 256)
        sA[idx] = f2bf(ablk[idx]);
    __syncthreads();

    const __bf16* arow = sA + (size_t)(lane & 15) * 1024;
    const __bf16* brow = WT + (size_t)(c0 + col) * 1024;

    v8f acc = {};
    for (int k0 = 0; k0 < 1024; k0 += 32) {
        __builtin_prefetch(brow + k0 + 256, 0, 1);
        v16bf a = a_from_bf(arow + k0 + hi * 16, arow + k0 + 8 + hi * 16);
        v16bf b = *(const v16bf*)(brow + k0 + hi * 16);
        acc = wmma_bf16(a, b, acc);
    }
    float bv = bias[c0 + col];
#pragma unroll
    for (int i = 0; i < 8; ++i)
        outbf[(size_t)(n0 + i + hi * 8) * 1024 + c0 + col] = f2bf(acc[i] + bv);
}

// ================= fused attention per (n-tile, group) =================
__global__ void __launch_bounds__(256)
k_att(const __bf16* __restrict__ qb, const __bf16* __restrict__ kb,
      const __bf16* __restrict__ logpos, const __bf16* __restrict__ VT,
      __bf16* __restrict__ ot) {
    __shared__ __align__(32) unsigned char smem[16 * 1024 * 4];  // 64 KB
    float*  s    = (float*)smem;           // f32 scores
    __bf16* s_bf = (__bf16*)smem;          // packed bf16 probs (first 32 KB)

    int w = threadIdx.x >> 5, lane = threadIdx.x & 31;
    int hi = lane >> 4, col = lane & 15;
    int n0 = blockIdx.x * 16;
    int g  = blockIdx.y;

    // ---- phase 1: scores = QK^T / 8 + logpos ----
    const __bf16* qrow = qb + (size_t)(n0 + col) * 1024 + g * 64;
    v16bf a0 = a_from_bf(qrow + hi * 16,      qrow + 8 + hi * 16);
    v16bf a1 = a_from_bf(qrow + 32 + hi * 16, qrow + 40 + hi * 16);

    for (int mt = w; mt < 64; mt += 8) {
        int m0 = mt * 16;
        const __bf16* krow = kb + (size_t)(m0 + col) * 1024 + g * 64;
        v16bf b0 = *(const v16bf*)(krow + hi * 16);
        v16bf b1 = *(const v16bf*)(krow + 32 + hi * 16);
        v8f c = {};
        c = wmma_bf16(a0, b0, c);
        c = wmma_bf16(a1, b1, c);
#pragma unroll
        for (int i = 0; i < 8; ++i) {
            int row = i + hi * 8;
            float lp = bf2f(logpos[((size_t)(n0 + row) * 16 + g) * 1024 + m0 + col]);
            s[row * 1024 + m0 + col] = c[i] * 0.125f + lp;
        }
    }
    __syncthreads();

    // ---- phase 2: row softmax; keep exp in regs, write back packed bf16 ----
    int r0 = w * 2, r1 = r0 + 1;
    const float* sr0 = s + r0 * 1024;
    const float* sr1 = s + r1 * 1024;
    float ev0[32], ev1[32];
    float mx0 = -3.0e38f, mx1 = -3.0e38f;
#pragma unroll
    for (int j = 0; j < 32; ++j) {
        ev0[j] = sr0[lane + j * 32];
        ev1[j] = sr1[lane + j * 32];
        mx0 = fmaxf(mx0, ev0[j]);
        mx1 = fmaxf(mx1, ev1[j]);
    }
#pragma unroll
    for (int o = 16; o > 0; o >>= 1) {
        mx0 = fmaxf(mx0, __shfl_xor(mx0, o, 32));
        mx1 = fmaxf(mx1, __shfl_xor(mx1, o, 32));
    }
    float sum0 = 0.f, sum1 = 0.f;
#pragma unroll
    for (int j = 0; j < 32; ++j) {
        ev0[j] = __expf(ev0[j] - mx0); sum0 += ev0[j];
        ev1[j] = __expf(ev1[j] - mx1); sum1 += ev1[j];
    }
#pragma unroll
    for (int o = 16; o > 0; o >>= 1) {
        sum0 += __shfl_xor(sum0, o, 32);
        sum1 += __shfl_xor(sum1, o, 32);
    }
    float inv0 = 1.0f / sum0, inv1 = 1.0f / sum1;
    __syncthreads();   // all f32 reads done before bf16 overwrite
#pragma unroll
    for (int j = 0; j < 32; ++j) {
        s_bf[r0 * 1024 + lane + j * 32] = f2bf(ev0[j] * inv0);
        s_bf[r1 * 1024 + lane + j * 32] = f2bf(ev1[j] * inv1);
    }
    __syncthreads();

    // ---- phase 3: output_t = P @ V  (A shared across 8 f-tiles per k-chunk) ----
    v8f acc[8];
    v8f z = {};
#pragma unroll
    for (int j = 0; j < 8; ++j) acc[j] = z;

    const __bf16* pr = s_bf + (size_t)(lane & 15) * 1024;
    for (int k0 = 0; k0 < 1024; k0 += 32) {
        __builtin_prefetch(VT + (size_t)(w * 16 + col) * 1024 + k0 + 256, 0, 1);
        v16bf a = a_from_bf(pr + k0 + hi * 16, pr + k0 + 8 + hi * 16);
#pragma unroll
        for (int j = 0; j < 8; ++j) {
            int f0 = (w + 8 * j) * 16;
            v16bf b = *(const v16bf*)(VT + (size_t)(f0 + col) * 1024 + k0 + hi * 16);
            acc[j] = wmma_bf16(a, b, acc[j]);
        }
    }
#pragma unroll
    for (int j = 0; j < 8; ++j) {
        int f0 = (w + 8 * j) * 16;
#pragma unroll
        for (int i = 0; i < 8; ++i)
            ot[(size_t)(n0 + i + hi * 8) * 16384 + (size_t)g * 1024 + f0 + col] =
                f2bf(acc[j][i]);
    }
}

// ================= grouped 1x1 conv: out = output_t @ W_conv^T + b =================
__global__ void __launch_bounds__(128)
k_conv(const __bf16* __restrict__ ot, const __bf16* __restrict__ Wc,
       const float* __restrict__ bconv, float* __restrict__ out) {
    int w = threadIdx.x >> 5, lane = threadIdx.x & 31;
    int hi = lane >> 4, col = lane & 15;
    int n0 = blockIdx.x * 16;
    int g  = blockIdx.y;
    int o0 = w * 16;

    const __bf16* arow = ot + (size_t)(n0 + col) * 16384 + (size_t)g * 1024;
    const __bf16* brow = Wc + ((size_t)g * 64 + o0 + col) * 1024;

    v8f acc = {};
    for (int k0 = 0; k0 < 1024; k0 += 32) {
        __builtin_prefetch(brow + k0 + 256, 0, 1);
        v16bf a = a_from_bf(arow + k0 + hi * 16, arow + k0 + 8 + hi * 16);
        v16bf b = *(const v16bf*)(brow + k0 + hi * 16);
        acc = wmma_bf16(a, b, acc);
    }
    float bv = bconv[g * 64 + o0 + col];
#pragma unroll
    for (int i = 0; i < 8; ++i)
        out[(size_t)(n0 + i + hi * 8) * 1024 + g * 64 + o0 + col] = acc[i] + bv;
}

// ================= host launcher =================
extern "C" void kernel_launch(void* const* d_in, const int* in_sizes, int n_in,
                              void* d_out, int out_size, void* d_ws, size_t ws_size,
                              hipStream_t stream) {
    const float* roi  = (const float*)d_in[0];
    const float* pe   = (const float*)d_in[1];
    const float* Wpos = (const float*)d_in[2];
    const float* bpos = (const float*)d_in[3];
    const float* Wq   = (const float*)d_in[4];
    const float* bq   = (const float*)d_in[5];
    const float* Wk   = (const float*)d_in[6];
    const float* bk   = (const float*)d_in[7];
    const float* Wc   = (const float*)d_in[8];
    const float* bc   = (const float*)d_in[9];
    float* out = (float*)d_out;

    char* ws = (char*)d_ws;
    const size_t MB = 1u << 20;
    __bf16* WqT    = (__bf16*)(ws + 0 * MB);    // 2 MB
    __bf16* WkT    = (__bf16*)(ws + 2 * MB);    // 2 MB
    __bf16* VT     = (__bf16*)(ws + 4 * MB);    // 2 MB
    __bf16* Wcbf   = (__bf16*)(ws + 6 * MB);    // 2 MB
    __bf16* WposT  = (__bf16*)(ws + 8 * MB);    // 2 KB
    __bf16* qbf    = (__bf16*)(ws + 9 * MB);    // 2 MB
    __bf16* kbf    = (__bf16*)(ws + 11 * MB);   // 2 MB
    __bf16* lpos   = (__bf16*)(ws + 13 * MB);   // 32 MB (N*G*M bf16)
    __bf16* otbf   = (__bf16*)(ws + 45 * MB);   // 32 MB (N*G*F bf16)

    // prep
    k_transpose_cvt<<<4096, 256, 0, stream>>>(Wq,   WqT, 1024, 1024);
    k_transpose_cvt<<<4096, 256, 0, stream>>>(Wk,   WkT, 1024, 1024);
    k_transpose_cvt<<<4096, 256, 0, stream>>>(roi,  VT,  1024, 1024);
    k_transpose_cvt<<<4,    256, 0, stream>>>(Wpos, WposT, 64, 16);
    k_copy_cvt<<<4096, 256, 0, stream>>>(Wc, Wcbf, 16 * 64 * 1024);

    // position MLP (reads the 256 MB tensor once)
    k_pos<<<8192, 256, 0, stream>>>(pe, WposT, bpos, lpos);

    // q / k projections
    k_proj<<<dim3(64, 8), 256, 0, stream>>>(roi, WqT, bq, qbf);
    k_proj<<<dim3(64, 8), 256, 0, stream>>>(roi, WkT, bk, kbf);

    // fused attention (QK^T + softmax + PV)
    k_att<<<dim3(64, 16), 256, 0, stream>>>(qbf, kbf, lpos, VT, otbf);

    // grouped conv + bias
    k_conv<<<dim3(64, 16), 128, 0, stream>>>(otbf, Wcbf, bc, out);
}
